// FC_Attention_42356967473257
// MI455X (gfx1250) — matrix operand order repaired and checked by reading, not compiled, not run on hardware
//
#include <hip/hip_runtime.h>

typedef unsigned short u16;
typedef unsigned int   u32;
typedef __attribute__((ext_vector_type(16))) __bf16 v16bf;
typedef __attribute__((ext_vector_type(8)))  float  v8f;

#define HWN   1024       // 32*32
#define NB    4          // batch
#define DQ    512
#define DV    256

// ---------- bf16 helpers ----------
__device__ __forceinline__ float bf2f(u16 h) {
  u32 u = ((u32)h) << 16;
  return __builtin_bit_cast(float, u);
}
__device__ __forceinline__ u16 f2bf(float f) {
  u32 u = __builtin_bit_cast(u32, f);
  u32 r = u + 0x7FFFu + ((u >> 16) & 1u);   // round-to-nearest-even
  return (u16)(r >> 16);
}

union FragBF { u32 u[8]; v16bf v; };

// ---------- elementwise / reduction kernels ----------

// NCHW f32 (256 ch) -> NHWC bf16
__global__ void k_nchw_to_nhwc_bf16(const float* __restrict__ src, u16* __restrict__ dst) {
  int idx = blockIdx.x * 256 + threadIdx.x;          // NHWC linear, 1M elems
  int c = idx & 255, p = (idx >> 8) & 1023, b = idx >> 18;
  dst[idx] = f2bf(src[(size_t)((b << 8) + c) * HWN + p]);
}

// f32 -> bf16 same layout
__global__ void k_cast_bf16(const float* __restrict__ src, u16* __restrict__ dst) {
  int idx = blockIdx.x * 256 + threadIdx.x;
  dst[idx] = f2bf(src[idx]);
}

// per-(b,c) spatial mean of x (NCHW)
__global__ void k_mean_x(const float* __restrict__ x, float* __restrict__ meanx) {
  __shared__ float sm[256];
  int bc = blockIdx.x, tid = threadIdx.x;
  const float* base = x + (size_t)bc * HWN;
  float a = base[tid] + base[tid + 256] + base[tid + 512] + base[tid + 768];
  sm[tid] = a; __syncthreads();
  for (int s = 128; s > 0; s >>= 1) { if (tid < s) sm[tid] += sm[tid + s]; __syncthreads(); }
  if (tid == 0) meanx[bc] = sm[0] * (1.f / 1024.f);
}

// bias[b, o] = sum_c meanx[b,c]*bias_w[o,c] + bias_b[o]
__global__ void k_bias(const float* __restrict__ meanx, const float* __restrict__ bw,
                       const float* __restrict__ bb, float* __restrict__ biasv) {
  int b = blockIdx.x, o = threadIdx.x;
  const float* mrow = meanx + b * 256;
  const float* wrow = bw + o * 256;
  float a = bb[o];
  #pragma unroll 4
  for (int c = 0; c < 256; ++c) a += mrow[c] * wrow[c];
  biasv[b * 256 + o] = a;
}

// OIHW f32 weights -> [tap][O][I] bf16
__global__ void k_w_cast(const float* __restrict__ w, u16* __restrict__ out, int Cout, int KS2) {
  int idx = blockIdx.x * 256 + threadIdx.x;      // = (t*Cout + o)*256 + c
  if (idx >= Cout * 256 * KS2) return;
  int c = idx & 255;
  int o = (idx >> 8) % Cout;
  int t = idx / (Cout * 256);
  out[idx] = f2bf(w[(size_t)(o * 256 + c) * KS2 + t]);
}

// bilinear sum-pool coefficient (32 -> 3, align_corners=False)
__device__ __forceinline__ float mcoef(int p, int i) {
  const float scale = 32.f / 3.f;
  float src = fmaxf((p + 0.5f) * scale - 0.5f, 0.f);
  int i0 = (int)floorf(src);
  float frac = src - (float)i0;
  int i0c = i0 > 31 ? 31 : i0;
  int i1c = (i0 + 1) > 31 ? 31 : (i0 + 1);
  float w = 0.f;
  if (i == i0c) w += 1.f - frac;
  if (i == i1c) w += frac;
  return w;
}

__global__ void k_pool_init(float* __restrict__ wpool) {
  int idx = blockIdx.x * 256 + threadIdx.x;     // 9*1024
  if (idx >= 9 * 1024) return;
  int pq = idx >> 10, hw = idx & 1023;
  int pp = pq / 3, qq = pq % 3;
  int y = hw >> 5, x = hw & 31;
  wpool[idx] = mcoef(pp, y) * mcoef(qq, x);
}

// global sums of relu^2 and relu^4
__global__ void k_reduce_phi(const float* __restrict__ src, int n,
                             float* __restrict__ s2, float* __restrict__ s4) {
  __shared__ float sm2[256], sm4[256];
  int tid = threadIdx.x;
  float a2 = 0.f, a4 = 0.f;
  for (int i = blockIdx.x * 256 + tid; i < n; i += gridDim.x * 256) {
    float r = fmaxf(src[i], 0.f);
    float p = r * r;
    a2 += p; a4 += p * p;
  }
  sm2[tid] = a2; sm4[tid] = a4; __syncthreads();
  for (int s = 128; s > 0; s >>= 1) {
    if (tid < s) { sm2[tid] += sm2[tid + s]; sm4[tid] += sm4[tid + s]; }
    __syncthreads();
  }
  if (tid == 0) { atomicAdd(s2, sm2[0]); atomicAdd(s4, sm4[0]); }
}

// phi(Q): NCHW f32 -> NHWC bf16 (scaled relu^2)
__global__ void k_phi_cast_nhwc(const float* __restrict__ src, const float* __restrict__ s,
                                u16* __restrict__ dst) {
  int idx = blockIdx.x * 256 + threadIdx.x;      // over 4*512*1024, NCHW linear
  float scale = sqrtf(s[0]) / sqrtf(fmaxf(s[1], 1e-30f));
  int p = idx & 1023;
  int ch = (idx >> 10) & 511;
  int b = idx >> 19;
  float r = fmaxf(src[idx], 0.f);
  dst[(size_t)((b << 10) + p) * DQ + ch] = f2bf(scale * r * r);
}

// phi(K): NCHW f32 -> NCHW bf16 (scaled relu^2)
__global__ void k_phi_cast_nchw(const float* __restrict__ src, const float* __restrict__ s,
                                u16* __restrict__ dst) {
  int idx = blockIdx.x * 256 + threadIdx.x;
  float scale = sqrtf(s[0]) / sqrtf(fmaxf(s[1], 1e-30f));
  float r = fmaxf(src[idx], 0.f);
  dst[idx] = f2bf(scale * r * r);
}

// ---------- WMMA kernels ----------

#define PIX_STRIDE 264   // 256 ch + 8 u16 pad (16B) -> 4-bank skew per pixel row

// Generic 3x3/1x1 conv, Cin=256, implicit GEMM.
// 4 waves/block: block computes 64(Cout) x 16(pixel); x-tile staged in LDS once
// per tap via CDNA5 async global->LDS copies (ASYNCcnt) and shared by all waves.
// wb: [tap][Cout][256] bf16, xb: NHWC bf16 (256 ch), out: NCHW f32 (+bias per Cout)
__global__ __launch_bounds__(128)
void k_conv_wmma(const u16* __restrict__ wb, const u16* __restrict__ xb,
                 const float* __restrict__ bias, float* __restrict__ out,
                 int Cout, int KS, int pad) {
  __shared__ __align__(16) u16 smB[16 * PIX_STRIDE];   // ~8.25 KB
  int tid  = threadIdx.x;
  int wave = tid >> 5, lane = tid & 31;
  int l = lane & 15, hf = lane >> 4;
  int m0 = blockIdx.x * 64 + wave * 16;
  int n0 = blockIdx.y * 16;
  int b  = blockIdx.z;

  // staging role: thread covers pixel lp, 4 of its 32 16-byte segments
  int lp   = tid >> 3;            // 0..15 local pixel
  int seg0 = tid & 7;             // segment phase
  int sp   = n0 + lp, spy = sp >> 5, spx = sp & 31;

  // compute role: lane's output pixel (same for both half-lanes)
  int p = n0 + l, py = p >> 5, px = p & 31;

  v8f acc = {};
  int nt = KS * KS;
  for (int tap = 0; tap < nt; ++tap) {
    int dy = tap / KS - pad, dx = tap % KS - pad;

    __syncthreads();   // previous tap's readers done before overwrite
    {
      int yy = spy + dy, xx = spx + dx;
      bool svalid = ((unsigned)yy < 32u) && ((unsigned)xx < 32u);
      u16* ldst = &smB[lp * PIX_STRIDE];
      if (svalid) {
        const u16* gsrc = xb + (size_t)((b << 10) + yy * 32 + xx) * 256;
        #pragma unroll
        for (int j = 0; j < 4; ++j) {
          int seg = seg0 + 8 * j;                       // 16B segment of this pixel row
          unsigned la = (unsigned)(uintptr_t)(ldst) + seg * 16;
          unsigned long long ga = (unsigned long long)(uintptr_t)(gsrc + seg * 8);
          asm volatile("global_load_async_to_lds_b128 %0, %1, off"
                       :: "v"(la), "v"(ga) : "memory");
        }
      } else {
        uint4 z = {0u, 0u, 0u, 0u};
        #pragma unroll
        for (int j = 0; j < 4; ++j) {
          int seg = seg0 + 8 * j;
          *(uint4*)((char*)ldst + seg * 16) = z;
        }
      }
    }
    asm volatile("s_wait_asynccnt 0x0" ::: "memory");
    __syncthreads();

    const u16* wrow = wb + (size_t)(tap * Cout + m0 + l) * 256;
    const u16* brow = &smB[l * PIX_STRIDE];
    for (int c0 = 0; c0 < 256; c0 += 32) {
      FragBF A, B;
      #pragma unroll
      for (int v = 0; v < 8; ++v) {            // A 16x32: K = (v<4?0:16) + hf*8 + (v&3)*2
        int koff = ((v < 4) ? 0 : 16) + hf * 8 + (v & 3) * 2;
        A.u[v] = *(const u32*)(wrow + c0 + koff);
      }
      __builtin_prefetch(wrow + c0 + 64, 0, 1);
      #pragma unroll
      for (int v = 0; v < 8; ++v) {            // B 32x16 from LDS: K = (v<4?0:16) + hf*8 + (v&3)*2
        int koff = ((v < 4) ? 0 : 16) + hf * 8 + (v & 3) * 2;
        B.u[v] = *(const u32*)(brow + c0 + koff);
      }
      acc = __builtin_amdgcn_wmma_f32_16x16x32_bf16(false, A.v, false, B.v,
                                                    (short)0, acc, false, false);
    }
  }
  #pragma unroll
  for (int r = 0; r < 8; ++r) {                // D: M = r + 8*hf, N = l
    int m = m0 + r + 8 * hf;
    out[(size_t)(b * Cout + m) * HWN + (n0 + l)] = acc[r] + bias[m];
  }
}

// KV pooled outer product: per (g, pq, tile): D(16kk x 16vv) = sum_hw phiK*w_pq x V
// phik: [b][512][1024] bf16, vb: [b][256][1024] bf16, wpool: [9][1024] f32
// filt out: [g][pq][vv(32)][kk(64)] bf16, clamped to +-32/9
__global__ __launch_bounds__(32)
void k_kv_wmma(const u16* __restrict__ phik, const u16* __restrict__ vb,
               const float* __restrict__ wpool, u16* __restrict__ filt) {
  int lane = threadIdx.x;
  int l = lane & 15, hf = lane >> 4;
  int t  = blockIdx.x;           // 0..7 : 4 kk-tiles x 2 vv-tiles
  int mi = t >> 1, ni = t & 1;
  int pq = blockIdx.y;
  int g  = blockIdx.z;
  int b = g >> 3, head = g & 7;
  int M0 = mi * 16, N0 = ni * 16;
  const u16* arow = phik + (size_t)(b * DQ + head * 64 + M0 + l) * HWN;
  const u16* brow = vb   + (size_t)(b * DV + head * 32 + N0 + l) * HWN;
  const float* wp = wpool + pq * HWN;
  v8f acc = {};
  for (int k0 = 0; k0 < HWN; k0 += 32) {
    FragBF A, B;
    #pragma unroll
    for (int v = 0; v < 8; ++v) {
      int koff = ((v < 4) ? 0 : 16) + hf * 8 + (v & 3) * 2;
      int hw = k0 + koff;
      u32 pr = *(const u32*)(arow + hw);
      float f0 = bf2f((u16)(pr & 0xFFFFu)) * wp[hw];
      float f1 = bf2f((u16)(pr >> 16))     * wp[hw + 1];
      A.u[v] = (u32)f2bf(f0) | ((u32)f2bf(f1) << 16);
    }
    #pragma unroll
    for (int v = 0; v < 8; ++v) {
      int koff = ((v < 4) ? 0 : 16) + hf * 8 + (v & 3) * 2;
      B.u[v] = *(const u32*)(brow + k0 + koff);
    }
    acc = __builtin_amdgcn_wmma_f32_16x16x32_bf16(false, A.v, false, B.v,
                                                  (short)0, acc, false, false);
  }
  const float amax = 32.f / 9.f;
  #pragma unroll
  for (int r = 0; r < 8; ++r) {
    int kk = M0 + r + 8 * hf;
    int vv = N0 + l;
    float val = fminf(fmaxf(acc[r], -amax), amax);
    filt[((size_t)(g * 9 + pq) * 32 + vv) * 64 + kk] = f2bf(val);
  }
}

// Dynamic grouped conv: per group g, out(vv=32) = filt * phiQ over (kk=64, 9 taps)
// filt: [g][tap][vv][kk] bf16, phiq: NHWC bf16 (512 ch), qkv out: NCHW f32 + biasv
__global__ __launch_bounds__(32)
void k_qkv_wmma(const u16* __restrict__ filt, const u16* __restrict__ phiq,
                const float* __restrict__ biasv, float* __restrict__ qkv) {
  int lane = threadIdx.x;
  int l = lane & 15, hf = lane >> 4;
  int M0 = blockIdx.x * 16;          // vv tile
  int N0 = blockIdx.y * 16;          // pixel tile
  int g  = blockIdx.z;
  int b = g >> 3, head = g & 7;
  int p = N0 + l, py = p >> 5, px = p & 31;
  v8f acc = {};
  for (int tap = 0; tap < 9; ++tap) {
    int dy = tap / 3 - 1, dx = tap % 3 - 1;
    int yy = py + dy, xx = px + dx;
    bool valid = ((unsigned)yy < 32u) && ((unsigned)xx < 32u);
    int pix = valid ? (yy * 32 + xx) : 0;
    const u16* bbase = phiq + (size_t)((b << 10) + pix) * DQ + head * 64;
    const u16* abase = filt + ((size_t)(g * 9 + tap) * 32 + M0 + l) * 64;
    for (int c0 = 0; c0 < 64; c0 += 32) {
      FragBF A, B;
      #pragma unroll
      for (int v = 0; v < 8; ++v) {
        int koff = ((v < 4) ? 0 : 16) + hf * 8 + (v & 3) * 2;
        A.u[v] = *(const u32*)(abase + c0 + koff);
      }
      #pragma unroll
      for (int v = 0; v < 8; ++v) {
        int koff = ((v < 4) ? 0 : 16) + hf * 8 + (v & 3) * 2;
        B.u[v] = valid ? *(const u32*)(bbase + c0 + koff) : 0u;
      }
      acc = __builtin_amdgcn_wmma_f32_16x16x32_bf16(false, A.v, false, B.v,
                                                    (short)0, acc, false, false);
    }
  }
  #pragma unroll
  for (int r = 0; r < 8; ++r) {
    int vv = M0 + r + 8 * hf;
    qkv[(size_t)(b * DV + head * 32 + vv) * HWN + (N0 + l)] = acc[r] + biasv[g * 32 + vv];
  }
}

// ---------- LayerNorm ----------
__global__ void k_ln_reduce(const float* __restrict__ src,
                            float* __restrict__ sums, float* __restrict__ sqs) {
  __shared__ float s1[256], s2[256];
  int b = blockIdx.y, tid = threadIdx.x;
  const float* base = src + (size_t)b * 262144;
  float a = 0.f, q = 0.f;
  for (int i = blockIdx.x * 256 + tid; i < 262144; i += gridDim.x * 256) {
    float v = base[i]; a += v; q += v * v;
  }
  s1[tid] = a; s2[tid] = q; __syncthreads();
  for (int s = 128; s > 0; s >>= 1) {
    if (tid < s) { s1[tid] += s1[tid + s]; s2[tid] += s2[tid + s]; }
    __syncthreads();
  }
  if (tid == 0) { atomicAdd(&sums[b], s1[0]); atomicAdd(&sqs[b], s2[0]); }
}

__global__ void k_ln_final(const float* __restrict__ src, const float* __restrict__ scal,
                           float* __restrict__ out) {
  int idx = blockIdx.x * 256 + threadIdx.x;
  int b = idx >> 18;
  float mean = scal[4 + b] * (1.f / 262144.f);
  float var  = scal[8 + b] * (1.f / 262144.f) - mean * mean;
  out[idx] = (src[idx] - mean) * rsqrtf(var + 1e-5f);
}

// ---------- launch ----------
extern "C" void kernel_launch(void* const* d_in, const int* in_sizes, int n_in,
                              void* d_out, int out_size, void* d_ws, size_t ws_size,
                              hipStream_t stream) {
  const float* x      = (const float*)d_in[0];
  const float* q_w    = (const float*)d_in[1];
  const float* q_b    = (const float*)d_in[2];
  const float* k_w    = (const float*)d_in[3];
  const float* k_b    = (const float*)d_in[4];
  const float* v_w    = (const float*)d_in[5];
  const float* v_b    = (const float*)d_in[6];
  const float* bias_w = (const float*)d_in[7];
  const float* bias_b = (const float*)d_in[8];
  const float* hu_w   = (const float*)d_in[9];
  const float* hu_b   = (const float*)d_in[10];
  float* out = (float*)d_out;

  char* ws = (char*)d_ws;
  const size_t MB = 1u << 20;
  u16*   xb    = (u16*)(ws + 0);                 // 2 MB  x NHWC bf16
  u16*   wqb   = (u16*)(ws + 2 * MB);            // 256 KB
  u16*   wkb   = (u16*)(ws + 3 * MB);            // 2.25 MB
  u16*   wvb   = (u16*)(ws + 6 * MB);            // 1.125 MB
  u16*   whub  = (u16*)(ws + 8 * MB);            // 1.125 MB
  float* wpool = (float*)(ws + 10 * MB);         // 36 KB
  float* meanx = (float*)(ws + 10 * MB + 64 * 1024);
  float* biasv = (float*)(ws + 10 * MB + 128 * 1024);
  float* scal  = (float*)(ws + 10 * MB + 192 * 1024);  // [0..1]=Q s2/s4 [2..3]=K [4..7]=LN sum [8..11]=LN sq
  float* Qf    = (float*)(ws + 11 * MB);         // 8 MB
  float* Kf    = (float*)(ws + 19 * MB);         // 8 MB
  float* Vf    = (float*)(ws + 27 * MB);         // 4 MB
  u16*   phiQ  = (u16*)(ws + 31 * MB);           // 4 MB NHWC
  u16*   phiK  = (u16*)(ws + 35 * MB);           // 4 MB NCHW
  u16*   Vb    = (u16*)(ws + 39 * MB);           // 2 MB NCHW
  u16*   filtB = (u16*)(ws + 41 * MB);           // 1.125 MB
  float* QKVf  = (float*)(ws + 43 * MB);         // 4 MB
  u16*   QKVb  = (u16*)(ws + 47 * MB);           // 2 MB NHWC
  float* OUTF  = (float*)(ws + 49 * MB);         // 4 MB

  hipMemsetAsync(scal, 0, 64 * sizeof(float), stream);

  // prep
  k_nchw_to_nhwc_bf16<<<4096, 256, 0, stream>>>(x, xb);
  k_mean_x<<<NB * 256, 256, 0, stream>>>(x, meanx);
  k_bias<<<NB, 256, 0, stream>>>(meanx, bias_w, bias_b, biasv);
  k_w_cast<<<512, 256, 0, stream>>>(q_w, wqb, 512, 1);
  k_w_cast<<<4608, 256, 0, stream>>>(k_w, wkb, 512, 9);
  k_w_cast<<<2304, 256, 0, stream>>>(v_w, wvb, 256, 9);
  k_w_cast<<<2304, 256, 0, stream>>>(hu_w, whub, 256, 9);
  k_pool_init<<<36, 256, 0, stream>>>(wpool);

  // Q/K/V convs (WMMA implicit GEMM, async-LDS staged x tiles)
  k_conv_wmma<<<dim3(8, 64, NB), 128, 0, stream>>>(wqb, xb, q_b, Qf, 512, 1, 0);
  k_conv_wmma<<<dim3(8, 64, NB), 128, 0, stream>>>(wkb, xb, k_b, Kf, 512, 3, 1);
  k_conv_wmma<<<dim3(4, 64, NB), 128, 0, stream>>>(wvb, xb, v_b, Vf, 256, 3, 1);

  // phi norms + casts
  k_reduce_phi<<<1024, 256, 0, stream>>>(Qf, NB * DQ * HWN, &scal[0], &scal[1]);
  k_reduce_phi<<<1024, 256, 0, stream>>>(Kf, NB * DQ * HWN, &scal[2], &scal[3]);
  k_phi_cast_nhwc<<<8192, 256, 0, stream>>>(Qf, &scal[0], phiQ);
  k_phi_cast_nchw<<<8192, 256, 0, stream>>>(Kf, &scal[2], phiK);
  k_cast_bf16<<<4096, 256, 0, stream>>>(Vf, Vb);

  // pooled KV filters (WMMA) and dynamic grouped conv (WMMA)
  k_kv_wmma<<<dim3(8, 9, 32), 32, 0, stream>>>(phiK, Vb, wpool, filtB);
  k_qkv_wmma<<<dim3(2, 64, 32), 32, 0, stream>>>(filtB, phiQ, biasv, QKVf);

  // head-unification conv (WMMA) + LayerNorm
  k_nchw_to_nhwc_bf16<<<4096, 256, 0, stream>>>(QKVf, QKVb);
  k_conv_wmma<<<dim3(4, 64, NB), 128, 0, stream>>>(whub, QKVb, hu_b, OUTF, 256, 3, 1);
  k_ln_reduce<<<dim3(64, NB), 256, 0, stream>>>(OUTF, &scal[4], &scal[8]);
  k_ln_final<<<4096, 256, 0, stream>>>(OUTF, scal, out);
}
